// EfficientMemoryGEMMWithSoftmax_20512763806070
// MI455X (gfx1250) — matrix operand order repaired and checked by reading, not compiled, not run on hardware
//
#include <hip/hip_runtime.h>

// ---- shapes (fixed by the reference) ----
#define S_DIM 1024
#define D_DIM 64
#define BH    32            // B*H = 2*16
#define MTILE 128           // rows of x1 per workgroup (8 waves * 16)
#define LDS_ROW_U32 516     // 512 data uints (1024 f16) + 4 pad uints per n-row
#define LDS_BYTES (D_DIM * LDS_ROW_U32 * 4)   // 132,096 B  (<= 320 KB/WG on CDNA5)

typedef __attribute__((ext_vector_type(16))) _Float16 v16h;
typedef __attribute__((ext_vector_type(8)))  float    v8f;

static __device__ __forceinline__ unsigned int pack_f16(float lo, float hi) {
    union { _Float16 h[2]; unsigned int u; } p;
    p.h[0] = (_Float16)lo;
    p.h[1] = (_Float16)hi;
    return p.u;
}

__global__ __launch_bounds__(256, 1)
void pv_gemm_f16_wmma(const float* __restrict__ P,   // x1: [BH, S, S]
                      const float* __restrict__ V,   // x2: [BH, S, D]
                      float* __restrict__ O)         // out:[BH, S, D]
{
    // All of x2[bh] lives in LDS, transposed to [n][k] as packed f16 pairs.
    extern __shared__ unsigned int Bsh[];

    const int bh   = blockIdx.y;
    const int m0   = blockIdx.x * MTILE;
    const int tid  = threadIdx.x;
    const int lane = tid & 31;
    const int wave = tid >> 5;
    const int half = lane >> 4;       // 0: lanes 0-15, 1: lanes 16-31
    const int lm   = lane & 15;

    const float* A  = P + (size_t)bh * S_DIM * S_DIM;
    const float* Bm = V + (size_t)bh * S_DIM * D_DIM;
    float*       C  = O + (size_t)bh * S_DIM * D_DIM;

    // ---- stage ALL of x2[bh] (1024x64 f32) into LDS as f16, transposed ----
    // thread t: column n = t&63, K-quarter q = t>>6 (256 K-rows each)
    {
        const int sn = tid & 63;
        const int sq = tid >> 6;
        unsigned int* srow = Bsh + sn * LDS_ROW_U32 + sq * 128;
        const float*  gcol = Bm + (size_t)(sq * 256) * D_DIM + sn;
        #pragma unroll 4
        for (int i = 0; i < 128; ++i) {
            const float lo = gcol[(size_t)(2 * i)     * D_DIM];
            const float hi = gcol[(size_t)(2 * i + 1) * D_DIM];
            srow[i] = pack_f16(lo, hi);
        }
    }
    __syncthreads();   // one barrier total; K-loop below is barrier-free

    const int mrow   = m0 + wave * 16 + lm;   // x1 row this lane feeds into A-frag
    const int kbaseA = half * 8;              // A-frag sub-K offset per ISA layout

    v8f acc[4];
    #pragma unroll
    for (int nt = 0; nt < 4; ++nt) acc[nt] = (v8f){};

    const float* arow_base = A + (size_t)mrow * S_DIM + kbaseA;

    // ---- main loop: 32 K-steps of 16x16x32, no barriers ----
    for (int ks = 0; ks < S_DIM / 32; ++ks) {
        const float* arow = arow_base + ks * 32;

        // speculative prefetch of the next K-step of this x1 row
        __builtin_prefetch(arow + 32, 0, 1);

        // issue A loads (global, longest latency) first
        const float4 a0 = *(const float4*)(arow + 0);
        const float4 a1 = *(const float4*)(arow + 4);
        const float4 a2 = *(const float4*)(arow + 16);
        const float4 a3 = *(const float4*)(arow + 20);

        // issue ALL FOUR B-fragment loads into distinct registers
        union bfu { v16h v; uint4 q[2]; } bf[4];
        const int uoff = ks * 16 + half * 4;   // uint offset inside n-row
        #pragma unroll
        for (int nt = 0; nt < 4; ++nt) {
            const unsigned int* bp = &Bsh[(nt * 16 + lm) * LDS_ROW_U32 + uoff];
            bf[nt].q[0] = *(const uint4*)(bp);
            bf[nt].q[1] = *(const uint4*)(bp + 8);
        }

        // convert A while the ds loads are in flight
        union { v16h v; _Float16 h[16]; } af;
        af.h[0]  = (_Float16)a0.x;  af.h[1]  = (_Float16)a0.y;
        af.h[2]  = (_Float16)a0.z;  af.h[3]  = (_Float16)a0.w;
        af.h[4]  = (_Float16)a1.x;  af.h[5]  = (_Float16)a1.y;
        af.h[6]  = (_Float16)a1.z;  af.h[7]  = (_Float16)a1.w;
        af.h[8]  = (_Float16)a2.x;  af.h[9]  = (_Float16)a2.y;
        af.h[10] = (_Float16)a2.z;  af.h[11] = (_Float16)a2.w;
        af.h[12] = (_Float16)a3.x;  af.h[13] = (_Float16)a3.y;
        af.h[14] = (_Float16)a3.z;  af.h[15] = (_Float16)a3.w;

        // four back-to-back WMMAs (no inter-WMMA waits: distinct B and D regs)
        #pragma unroll
        for (int nt = 0; nt < 4; ++nt) {
            acc[nt] = __builtin_amdgcn_wmma_f32_16x16x32_f16(
                /*neg_a=*/false, af.v,
                /*neg_b=*/false, bf[nt].v,
                /*c_mod=*/(short)0, acc[nt],
                /*reuse_a=*/false, /*reuse_b=*/false);
        }
    }

    // ---- epilogue: C/D layout -> VGPR r holds M = half*8 + r, N = lm ----
    const int mbase = m0 + wave * 16 + half * 8;
    #pragma unroll
    for (int nt = 0; nt < 4; ++nt) {
        const int n = nt * 16 + lm;
        #pragma unroll
        for (int r = 0; r < 8; ++r) {
            C[(size_t)(mbase + r) * D_DIM + n] = acc[nt][r];
        }
    }
}

extern "C" void kernel_launch(void* const* d_in, const int* in_sizes, int n_in,
                              void* d_out, int out_size, void* d_ws, size_t ws_size,
                              hipStream_t stream) {
    const float* x1 = (const float*)d_in[0];   // [2,16,1024,1024] fp32
    const float* x2 = (const float*)d_in[1];   // [2,16,1024,64]   fp32
    float* out      = (float*)d_out;           // [2,16,1024,64]   fp32

    dim3 grid(S_DIM / MTILE, BH);   // (8, 32)
    dim3 block(256);                // 8 waves (wave32)
    pv_gemm_f16_wmma<<<grid, block, LDS_BYTES, stream>>>(x1, x2, out);
}
